// ResGCN_71425306132758
// MI455X (gfx1250) — matrix operand order repaired and checked by reading, not compiled, not run on hardware
//
#include <hip/hip_runtime.h>
#include <hip/hip_bf16.h>

typedef __attribute__((ext_vector_type(16))) _Float16 v16h;
typedef __attribute__((ext_vector_type(8)))  _Float16 v8h;
typedef __attribute__((ext_vector_type(8)))  float    v8f;

#define NN 100000
#define EE 800000
#define HH 128
#define GG 10
#define CC 10
#define EPSF 1e-5f

// ---------------------------------------------------------------- utility
__global__ void zero_f32(float* __restrict__ p, int n) {
    for (int i = blockIdx.x * blockDim.x + threadIdx.x; i < n; i += gridDim.x * blockDim.x)
        p[i] = 0.f;
}

__global__ void deg_init(float* __restrict__ d, int n) {
    for (int i = blockIdx.x * blockDim.x + threadIdx.x; i < n; i += gridDim.x * blockDim.x)
        d[i] = 1.0f;   // self-loop weight
}

__global__ void deg_edges(const int* __restrict__ ei, const float* __restrict__ ew,
                          float* __restrict__ d, int e) {
    for (int i = blockIdx.x * blockDim.x + threadIdx.x; i < e; i += gridDim.x * blockDim.x)
        atomicAdd(&d[ei[e + i]], ew[i]);          // col = ei[E + i]
}

__global__ void dinv_finalize(float* __restrict__ d, int n) {
    for (int i = blockIdx.x * blockDim.x + threadIdx.x; i < n; i += gridDim.x * blockDim.x) {
        float v = d[i];
        d[i] = (v > 0.f) ? rsqrtf(fmaxf(v, EPSF)) : 0.f;
    }
}

// ---------------------------------------------------------------- batchnorm
// Standalone stats pass (only needed for the external input x).
__global__ __launch_bounds__(128) void bn_stats(const float* __restrict__ h,
                                                float* __restrict__ stats, int n) {
    int f = threadIdx.x;
    float s = 0.f, s2 = 0.f;
    for (int r = blockIdx.x; r < n; r += gridDim.x) {
        float v = h[(size_t)r * HH + f];
        s += v; s2 += v * v;
    }
    atomicAdd(&stats[f], s);
    atomicAdd(&stats[HH + f], s2);
}

__global__ __launch_bounds__(128) void bn_apply_cast(const float* __restrict__ h,
        const float* __restrict__ stats, const float* __restrict__ w,
        const float* __restrict__ b, _Float16* __restrict__ out, int n, float invN) {
    int f = threadIdx.x;
    float mu  = stats[f] * invN;
    float var = stats[HH + f] * invN - mu * mu;
    float sc  = rsqrtf(var + EPSF) * w[f];
    float sh  = b[f] - mu * sc;
    for (int r = blockIdx.x; r < n; r += gridDim.x)
        out[(size_t)r * HH + f] = (_Float16)(h[(size_t)r * HH + f] * sc + sh);
}

// BN -> ReLU, and accumulate stats of the OUTPUT into statsOut (pre-zeroed).
__global__ __launch_bounds__(128) void bn_apply_relu_stats(const float* __restrict__ h,
        const float* __restrict__ statsIn, const float* __restrict__ w,
        const float* __restrict__ b, float* __restrict__ out,
        float* __restrict__ statsOut, int n, float invN) {
    int f = threadIdx.x;
    float mu  = statsIn[f] * invN;
    float var = statsIn[HH + f] * invN - mu * mu;
    float sc  = rsqrtf(var + EPSF) * w[f];
    float sh  = b[f] - mu * sc;
    float s = 0.f, s2 = 0.f;
    for (int r = blockIdx.x; r < n; r += gridDim.x) {
        float v = fmaxf(h[(size_t)r * HH + f] * sc + sh, 0.f);
        out[(size_t)r * HH + f] = v;
        s += v; s2 += v * v;
    }
    atomicAdd(&statsOut[f], s);
    atomicAdd(&statsOut[HH + f], s2);
}

__global__ void convert_w16(const float* __restrict__ W, _Float16* __restrict__ w16, int n) {
    for (int i = blockIdx.x * blockDim.x + threadIdx.x; i < n; i += gridDim.x * blockDim.x)
        w16[i] = (_Float16)W[i];
}

// ---------------------------------------------------------------- WMMA GEMM
// Y[N,128] = A16[N,128] x W16[128,128], f16 in / f32 accumulate.
// Epilogue also writes Acc = dinv[r]^2 * Y (self-loop term of the scatter).
__global__ __launch_bounds__(256) void gemm_n128(const _Float16* __restrict__ A,
        const _Float16* __restrict__ W, const float* __restrict__ dinv,
        float* __restrict__ Y, float* __restrict__ Acc, int n) {
    __shared__ _Float16 Wt[HH * HH];              // Wt[col*128 + k] (transposed)
    for (int i = threadIdx.x; i < HH * HH; i += 256) {
        int k = i >> 7, c = i & 127;
        Wt[c * HH + k] = W[i];
    }
    __syncthreads();

    const int wave  = threadIdx.x >> 5;
    const int lane  = threadIdx.x & 31;
    const int row0  = blockIdx.x * 128 + wave * 16;
    const int m     = lane & 15;
    const int lh    = lane >> 4;                  // lane-half select
    const int r     = row0 + m;
    const int rs    = (r < n) ? r : (n - 1);      // clamp loads; EXEC stays full

    v8f acc[8] = {};

    #pragma unroll
    for (int kt = 0; kt < 4; ++kt) {
        const int kb   = kt * 32;
        const int koff = lh * 8;
        // A fragment: 16-bit A 16x32 layout (two contiguous 16B chunks per lane)
        const _Float16* ap = A + (size_t)rs * HH + kb + koff;
        v8h alo = *(const v8h*)(ap);
        v8h ahi = *(const v8h*)(ap + 16);
        v16h a;
        #pragma unroll
        for (int j = 0; j < 8; ++j) { a[j] = alo[j]; a[8 + j] = ahi[j]; }

        #pragma unroll
        for (int nt = 0; nt < 8; ++nt) {
            // B fragment: 16-bit B 32x16 layout; contiguous in transposed LDS
            const int c = nt * 16 + m;
            const _Float16* bp = Wt + c * HH + kb + lh * 8;
            v8h blo = *(const v8h*)(bp);
            v8h bhi = *(const v8h*)(bp + 16);
            v16h bf;
            #pragma unroll
            for (int j = 0; j < 8; ++j) { bf[j] = blo[j]; bf[8 + j] = bhi[j]; }

            acc[nt] = __builtin_amdgcn_wmma_f32_16x16x32_f16(
                false, a, false, bf, (short)0, acc[nt], false, false);
        }
    }

    // self-loop coefficients for this lane's 8 output rows
    float d2[8];
    #pragma unroll
    for (int v = 0; v < 8; ++v) {
        int rv = row0 + v + lh * 8;
        float d = dinv[(rv < n) ? rv : 0];
        d2[v] = d * d;
    }

    // D layout: vgpr v -> row = row0 + v + 8*lh, col = nt*16 + (lane&15)
    #pragma unroll
    for (int nt = 0; nt < 8; ++nt) {
        const int c = nt * 16 + m;
        #pragma unroll
        for (int v = 0; v < 8; ++v) {
            int rr = row0 + v + lh * 8;
            if (rr < n) {
                float val = acc[nt][v];
                Y  [(size_t)rr * HH + c] = val;
                Acc[(size_t)rr * HH + c] = d2[v] * val;
            }
        }
    }
}

// ---------------------------------------------------------------- scatter
__global__ __launch_bounds__(256) void edge_scatter(const float* __restrict__ y,
        const int* __restrict__ ei, const float* __restrict__ ew,
        const float* __restrict__ dinv, float* __restrict__ acc, int e) {
    int f   = threadIdx.x & 127;
    int sub = threadIdx.x >> 7;                   // 2 edges per block iteration
    int step = gridDim.x * 2;
    for (int k = blockIdx.x * 2 + sub; k < e; k += step) {
        int kn = k + step;
        if (kn < e) {                             // prefetch next gather row
            int rn = ei[kn];
            __builtin_prefetch(&y[(size_t)rn * HH + f], 0, 0);
        }
        int rr = ei[k];
        int cc = ei[e + k];
        float coef = dinv[rr] * ew[k] * dinv[cc];
        atomicAdd(&acc[(size_t)cc * HH + f], coef * y[(size_t)rr * HH + f]);
    }
}

// bias + relu, and accumulate stats of the OUTPUT into statsOut (pre-zeroed).
__global__ __launch_bounds__(128) void bias_relu_stats(const float* __restrict__ acc,
        const float* __restrict__ b, float* __restrict__ out,
        float* __restrict__ statsOut, int n) {
    int f = threadIdx.x;
    float bf = b[f];
    float s = 0.f, s2 = 0.f;
    for (int r = blockIdx.x; r < n; r += gridDim.x) {
        float v = fmaxf(acc[(size_t)r * HH + f] + bf, 0.f);
        out[(size_t)r * HH + f] = v;
        s += v; s2 += v * v;
    }
    atomicAdd(&statsOut[f], s);
    atomicAdd(&statsOut[HH + f], s2);
}

// ---------------------------------------------------------------- head
__global__ __launch_bounds__(128) void pool_kernel(const float* __restrict__ h,
        const int* __restrict__ batch, float* __restrict__ g, int n) {
    __shared__ float gl[GG * HH];
    for (int i = threadIdx.x; i < GG * HH; i += 128) gl[i] = 0.f;
    __syncthreads();
    int f = threadIdx.x;
    for (int r = blockIdx.x; r < n; r += gridDim.x)
        gl[batch[r] * HH + f] += h[(size_t)r * HH + f];   // only thread f touches col f
    __syncthreads();
    for (int i = threadIdx.x; i < GG * HH; i += 128) atomicAdd(&g[i], gl[i]);
}

__global__ __launch_bounds__(128) void head_kernel(const float* __restrict__ g,
        const float* __restrict__ bw1, const float* __restrict__ bb1,
        const float* __restrict__ fcW, const float* __restrict__ fcb,
        const float* __restrict__ bw2, const float* __restrict__ bb2,
        const float* __restrict__ clW, const float* __restrict__ clb,
        float* __restrict__ out) {
    __shared__ float t0[GG * HH];
    __shared__ float t1[GG * HH];
    __shared__ float lg[GG * CC];
    __shared__ float rm[GG], rl[GG];
    int f = threadIdx.x;

    for (int r = 0; r < GG; ++r) t0[r * HH + f] = g[r * HH + f];
    // BN1 over G rows (per-feature, own column only)
    float s = 0.f, s2 = 0.f;
    for (int r = 0; r < GG; ++r) { float v = t0[r * HH + f]; s += v; s2 += v * v; }
    float mu = s * (1.f / GG), var = s2 * (1.f / GG) - mu * mu;
    float sc = rsqrtf(var + EPSF) * bw1[f], sh = bb1[f] - mu * sc;
    for (int r = 0; r < GG; ++r) t1[r * HH + f] = t0[r * HH + f] * sc + sh;
    __syncthreads();
    // FC + relu
    for (int r = 0; r < GG; ++r) {
        float a = fcb[f];
        for (int k = 0; k < HH; ++k) a += t1[r * HH + k] * fcW[k * HH + f];
        t0[r * HH + f] = fmaxf(a, 0.f);
    }
    __syncthreads();
    // BN2
    s = 0.f; s2 = 0.f;
    for (int r = 0; r < GG; ++r) { float v = t0[r * HH + f]; s += v; s2 += v * v; }
    mu = s * (1.f / GG); var = s2 * (1.f / GG) - mu * mu;
    sc = rsqrtf(var + EPSF) * bw2[f]; sh = bb2[f] - mu * sc;
    for (int r = 0; r < GG; ++r) t1[r * HH + f] = t0[r * HH + f] * sc + sh;
    __syncthreads();
    // classifier logits
    if (f < GG * CC) {
        int r = f / CC, c = f % CC;
        float a = clb[c];
        for (int k = 0; k < HH; ++k) a += t1[r * HH + k] * clW[k * CC + c];
        lg[f] = a;
    }
    __syncthreads();
    if (f < GG) {
        float m = -1e30f;
        for (int c = 0; c < CC; ++c) m = fmaxf(m, lg[f * CC + c]);
        float se = 0.f;
        for (int c = 0; c < CC; ++c) se += __expf(lg[f * CC + c] - m);
        rm[f] = m; rl[f] = logf(se);
    }
    __syncthreads();
    if (f < GG * CC) out[f] = lg[f] - rm[f / CC] - rl[f / CC];
}

// ---------------------------------------------------------------- launcher
extern "C" void kernel_launch(void* const* d_in, const int* in_sizes, int n_in,
                              void* d_out, int out_size, void* d_ws, size_t ws_size,
                              hipStream_t stream) {
    const float* x        = (const float*)d_in[0];
    const int*   ei       = (const int*)  d_in[1];
    const int*   batch    = (const int*)  d_in[2];
    const float* ew       = (const float*)d_in[3];
    const float* bnf_w    = (const float*)d_in[4];
    const float* bnf_b    = (const float*)d_in[5];
    const float* cf_W     = (const float*)d_in[6];
    const float* cf_b     = (const float*)d_in[7];
    const float* bns_w    = (const float*)d_in[8];
    const float* bns_b    = (const float*)d_in[9];
    const float* convs_W  = (const float*)d_in[10];
    const float* convs_b  = (const float*)d_in[11];
    const float* bnfc_w   = (const float*)d_in[12];
    const float* bnfc_b   = (const float*)d_in[13];
    const float* fc_W     = (const float*)d_in[14];
    const float* fc_b     = (const float*)d_in[15];
    const float* bnh_w    = (const float*)d_in[16];
    const float* bnh_b    = (const float*)d_in[17];
    const float* cl_W     = (const float*)d_in[18];
    const float* cl_b     = (const float*)d_in[19];
    float* out = (float*)d_out;

    // workspace carving (256B aligned)
    char* p = (char*)d_ws;
    auto alloc = [&](size_t bytes) { void* r = (void*)p; p += (bytes + 255) & ~(size_t)255; return r; };
    float*    dinv   = (float*)   alloc((size_t)NN * 4);
    float*    sx     = (float*)   alloc(2 * HH * 4);   // stats of x
    float*    sHI    = (float*)   alloc(2 * HH * 4);   // stats of hinit
    float*    sB1    = (float*)   alloc(2 * HH * 4);
    float*    sB2    = (float*)   alloc(2 * HH * 4);
    float*    sB3    = (float*)   alloc(2 * HH * 4);
    _Float16* w16f   = (_Float16*)alloc((size_t)HH * HH * 2);
    _Float16* w160   = (_Float16*)alloc((size_t)HH * HH * 2);
    _Float16* w161   = (_Float16*)alloc((size_t)HH * HH * 2);
    _Float16* w162   = (_Float16*)alloc((size_t)HH * HH * 2);
    _Float16* a16    = (_Float16*)alloc((size_t)NN * HH * 2);
    float*    ybuf   = (float*)   alloc((size_t)NN * HH * 4);
    float*    accbuf = (float*)   alloc((size_t)NN * HH * 4);
    float*    hinit  = (float*)   alloc((size_t)NN * HH * 4);
    float*    B1     = (float*)   alloc((size_t)NN * HH * 4);
    float*    B2     = (float*)   alloc((size_t)NN * HH * 4);
    float*    B3     = (float*)   alloc((size_t)NN * HH * 4);
    float*    gbuf   = (float*)   alloc((size_t)GG * HH * 4);

    const float invN = 1.0f / (float)NN;
    const int gemmBlocks = (NN + 127) / 128;

    // degree / symmetric normalization
    deg_init     <<<512, 256, 0, stream>>>(dinv, NN);
    deg_edges    <<<2048, 256, 0, stream>>>(ei, ew, dinv, EE);
    dinv_finalize<<<512, 256, 0, stream>>>(dinv, NN);

    // pre-convert all weight matrices to f16 once
    convert_w16<<<64, 256, 0, stream>>>(cf_W,                w16f, HH * HH);
    convert_w16<<<64, 256, 0, stream>>>(convs_W,             w160, HH * HH);
    convert_w16<<<64, 256, 0, stream>>>(convs_W + HH * HH,   w161, HH * HH);
    convert_w16<<<64, 256, 0, stream>>>(convs_W + 2*HH*HH,   w162, HH * HH);

    // stats of the external input x (everything else gets fused stats)
    zero_f32<<<1, 256, 0, stream>>>(sx, 2 * HH);
    bn_stats<<<512, 128, 0, stream>>>(x, sx, NN);

    auto conv_relu = [&](const float* hin, const float* statsIn,
                         const float* bnw, const float* bnb,
                         const _Float16* w16, const float* bias,
                         float* hout, float* statsOut) {
        zero_f32       <<<1, 256, 0, stream>>>(statsOut, 2 * HH);
        bn_apply_cast  <<<1024, 128, 0, stream>>>(hin, statsIn, bnw, bnb, a16, NN, invN);
        gemm_n128      <<<gemmBlocks, 256, 0, stream>>>(a16, w16, dinv, ybuf, accbuf, NN);
        edge_scatter   <<<4096, 256, 0, stream>>>(ybuf, ei, ew, dinv, accbuf, EE);
        bias_relu_stats<<<1024, 128, 0, stream>>>(accbuf, bias, hout, statsOut, NN);
    };
    auto bn_relu = [&](const float* hin, const float* statsIn,
                       const float* bnw, const float* bnb,
                       float* hout, float* statsOut) {
        zero_f32          <<<1, 256, 0, stream>>>(statsOut, 2 * HH);
        bn_apply_relu_stats<<<1024, 128, 0, stream>>>(hin, statsIn, bnw, bnb,
                                                      hout, statsOut, NN, invN);
    };
    auto head = [&](const float* hn, int off) {
        zero_f32   <<<8, 256, 0, stream>>>(gbuf, GG * HH);
        pool_kernel<<<1024, 128, 0, stream>>>(hn, batch, gbuf, NN);
        head_kernel<<<1, 128, 0, stream>>>(gbuf, bnfc_w, bnfc_b, fc_W, fc_b,
                                           bnh_w, bnh_b, cl_W, cl_b, out + off);
    };

    const float* bw0 = bns_w;           const float* bb0 = bns_b;
    const float* bw1 = bns_w + HH;      const float* bb1 = bns_b + HH;
    const float* bw2 = bns_w + 2 * HH;  const float* bb2 = bns_b + 2 * HH;
    const float* cb0 = convs_b;
    const float* cb1 = convs_b + HH;
    const float* cb2 = convs_b + 2 * HH;

    // feature conv: h_init = relu(conv(bn_feat(x)))
    conv_relu(x, sx, bnf_w, bnf_b, w16f, cf_b, hinit, sHI);

    // branch 1 (== branch 3); keep step-0 output (B1, sB1) for branch 2 reuse
    conv_relu(hinit, sHI, bw0, bb0, w160, cb0, B1, sB1);
    conv_relu(B1,    sB1, bw1, bb1, w161, cb1, B2, sB2);
    conv_relu(B2,    sB2, bw2, bb2, w162, cb2, B3, sB3);
    head(B3, 0);      // branch 1
    head(B3, 200);    // branch 3 (identical)

    // branch 2: h_(i=0) == B1 (stats sB1 still valid: B1 untouched)
    conv_relu(B1, sB1, bw0, bb0, w160, cb0, B2, sB2);   // h  after i=0
    conv_relu(B2, sB2, bw1, bb1, w161, cb1, B3, sB3);   // h_ i=1
    conv_relu(B3, sB3, bw1, bb1, w161, cb1, B2, sB2);   // h  i=1
    conv_relu(B2, sB2, bw2, bb2, w162, cb2, B3, sB3);   // h_ i=2
    conv_relu(B3, sB3, bw2, bb2, w162, cb2, B2, sB2);   // h  i=2
    head(B2, 100);    // branch 2

    // branch 4: BN -> ReLU only (hinit/sHI still valid); ping-pong stats slots
    bn_relu(hinit, sHI, bw0, bb0, B3, sB3);
    bn_relu(B3,    sB3, bw1, bb1, B3, sB1);   // in-place ok; stats slot differs
    bn_relu(B3,    sB1, bw2, bb2, B3, sB2);   // output stats unused
    head(B3, 300);    // branch 4
}
